// SNN_Perceptron_Equalizer_69526930587731
// MI455X (gfx1250) — matrix operand order repaired and checked by reading, not compiled, not run on hardware
//
#include <hip/hip_runtime.h>

// ---------------------------------------------------------------------------
// SNN LIF-recurrent scan for MI455X (gfx1250, wave32, WMMA).
//   T=1024 B=512 IN=64 HID=128 OUT=2
//   Fusions: W_comb = cell_W_in @ W_in  (removes one GEMM from the loop)
//            z_sum  = (sum_t z_t) @ W_out.T + T*b_out (removes W_out from loop)
//   Loop body per WG (16 batch rows, 4 waves x 32 hidden cols):
//     TDM prefetch of next x-tile (f16) into LDS double buffer,
//     GEMM1 cur = x @ W_comb.T  (2x v_wmma per N-tile),
//     GEMM2 rec = z_prev @ W_rec.T (4x v_wmma per N-tile),
//     elementwise LIF update, z -> LDS (double buffered), 1 barrier/step.
// ---------------------------------------------------------------------------

#define TT   1024
#define BB   512
#define INF  64
#define HID  128
#define TB   16            // batch rows per workgroup
#define NWG  (BB / TB)     // 32 workgroups
#define DTC  0.001f

typedef __attribute__((ext_vector_type(16))) _Float16 v16h;
typedef __attribute__((ext_vector_type(8)))  float    v8f;
typedef __attribute__((ext_vector_type(4)))  unsigned u32x4;
typedef __attribute__((ext_vector_type(8)))  int      i32x8;
typedef __attribute__((ext_vector_type(4)))  int      i32x4;
typedef __attribute__((ext_vector_type(4)))  float    f32x4;
typedef __attribute__((ext_vector_type(4)))  _Float16 h16x4;

union AF { v16h v; u32x4 q[2]; };

// A fragment (16x32 f16, MxK): lane holds row (lane&15); halves e -> K = 32c + 16*lhi + e
static __device__ __forceinline__ v16h load_fragA(const _Float16* p) {
  AF f;
  f.q[0] = *reinterpret_cast<const u32x4*>(p);
  f.q[1] = *reinterpret_cast<const u32x4*>(p + 8);
  return f.v;
}
// B fragment (32x16 f16, KxN): lane holds col (lane&15); halves e -> K = 32c + 16*lhi + e
// (B[k][n] = W[n][k] for D = A @ W^T, so lane reads row n of W, 16 contiguous halves)
static __device__ __forceinline__ v16h load_fragB(const _Float16* p) {
  AF f;
  f.q[0] = *reinterpret_cast<const u32x4*>(p);
  f.q[1] = *reinterpret_cast<const u32x4*>(p + 8);
  return f.v;
}

static __device__ __forceinline__ v8f wmma_f16(v16h a, v16h b, v8f c) {
  return __builtin_amdgcn_wmma_f32_16x16x32_f16(false, a, false, b, (short)0, c,
                                                false, false);
}

// Tensor Data Mover: async 2D tile load (16 rows x 64 f16) global -> LDS.
// D# per CDNA5 ISA (group0/group1 bitfields).
// 6-arg builtin form (amdgpu-toolchain / clang-23).
static __device__ __forceinline__ void tdm_load_tile(const _Float16* gsrc,
                                                     unsigned lds_byte_off) {
  unsigned long long ga = (unsigned long long)(size_t)gsrc;
  u32x4 g0 = { 1u,                                   // count=1, valid
               lds_byte_off,                         // lds_addr (bytes)
               (unsigned)(ga & 0xFFFFFFFFu),         // global_addr[31:0]
               (unsigned)((ga >> 32) & 0x01FFFFFFu)  // global_addr[56:32]
                 | (2u << 30) };                     // type=2 ("image")
  i32x8 g1 = { (1 << 16),       // data_size=1 -> 2-byte elements
               (INF << 16),     // tensor_dim0 = 64  (bits 79:48)
               (TB  << 16),     // tensor_dim1 = 16  (bits 111:80)
               (INF << 16),     // tile_dim0   = 64  (bits 127:112)
               TB,              // tile_dim1   = 16  (bits 143:128)
               INF,             // tensor_dim0_stride = 64 (bits 207:160)
               0, 0 };
  i32x4 gz4 = { 0, 0, 0, 0 };
  i32x8 gz8 = { 0, 0, 0, 0, 0, 0, 0, 0 };
  __builtin_amdgcn_tensor_load_to_lds(g0, g1, gz4, gz4, gz8, 0);
}

// --------------------------- kernel 1: x -> f16 ----------------------------
__global__ void convx_kernel(const float* __restrict__ x,
                             _Float16* __restrict__ x16) {
  size_t i = ((size_t)blockIdx.x * blockDim.x + threadIdx.x) * 4;
  f32x4 v = *reinterpret_cast<const f32x4*>(x + i);
  *reinterpret_cast<h16x4*>(x16 + i) = __builtin_convertvector(v, h16x4);
}

// ------------------- kernel 2: weight prep (tiny, 1 block) -----------------
__global__ void prep_kernel(const float* __restrict__ W_in,     // [HID][INF]
                            const float* __restrict__ cW_in,    // [HID][HID]
                            const float* __restrict__ cW_rec,   // [HID][HID]
                            _Float16* __restrict__ wcomb,       // [HID][INF]
                            _Float16* __restrict__ wrec) {      // [HID][HID]
  int h = threadIdx.x;                  // 0..127
  for (int k = 0; k < INF; ++k) {
    float acc = 0.0f;
    for (int j = 0; j < HID; ++j) acc += cW_in[h * HID + j] * W_in[j * INF + k];
    wcomb[h * INF + k] = (_Float16)acc;
  }
  for (int k = 0; k < HID; ++k) wrec[h * HID + k] = (_Float16)cW_rec[h * HID + k];
}

// --------------------------- kernel 3: the scan ----------------------------
__global__ __launch_bounds__(128, 1) void snn_scan_kernel(
    const _Float16* __restrict__ x16,     // [TT][BB][INF]
    const _Float16* __restrict__ wcomb,   // [HID][INF]
    const _Float16* __restrict__ wrec,    // [HID][HID]
    const float* __restrict__ W_out,      // [2][HID]
    const float* __restrict__ b_out,      // [2]
    const float* __restrict__ v_th,       // [HID]
    const float* __restrict__ v_leak_p,   // scalar
    const float* __restrict__ v_reset,    // [HID]
    const float* __restrict__ tau_mem_inv,// [HID]
    const float* __restrict__ tau_syn_inv,// [HID]
    float* __restrict__ out,              // [BB*2 + 1]
    float* __restrict__ spk_partial) {    // [NWG]
  __shared__ __align__(16) _Float16 sm_x[2 * TB * INF];  // 4 KB double-buffered
  __shared__ __align__(16) _Float16 sm_z[2 * TB * HID];  // 8 KB double-buffered
  __shared__ __align__(16) float    sm_zs[TB * HID];     // 8 KB spike sums

  const int tid  = threadIdx.x;
  const int wave = tid >> 5;
  const int lane = tid & 31;
  const int lrow = lane & 15;
  const int lhi  = lane >> 4;
  const int rb   = blockIdx.x * TB;

  // Per-column LIF params (col = hidden index = 32*wave + 16*n + lrow)
  float dtm[2], dts[2], vth[2], vrst[2];
  const float vleak = v_leak_p[0];
#pragma unroll
  for (int n = 0; n < 2; ++n) {
    int h   = 32 * wave + 16 * n + lrow;
    dtm[n]  = DTC * tau_mem_inv[h];
    dts[n]  = DTC * tau_syn_inv[h];
    vth[n]  = v_th[h];
    vrst[n] = v_reset[h];
  }

  // Loop-invariant B fragments kept in VGPRs.
  v16h bcomb[2][2], brec[2][4];
#pragma unroll
  for (int n = 0; n < 2; ++n) {
    int wn = 32 * wave + 16 * n + lrow;
    const _Float16* cr = wcomb + wn * INF;
    const _Float16* rr = wrec  + wn * HID;
#pragma unroll
    for (int c = 0; c < 2; ++c) bcomb[n][c] = load_fragB(cr + 32 * c + 16 * lhi);
#pragma unroll
    for (int c = 0; c < 4; ++c) brec[n][c]  = load_fragB(rr + 32 * c + 16 * lhi);
  }

  // LIF state in registers, WMMA C/D layout: elem r -> (M=r+8*lhi, N=col)
  float Vst[2][8], Ist[2][8], Zac[2][8];
#pragma unroll
  for (int n = 0; n < 2; ++n)
#pragma unroll
    for (int r = 0; r < 8; ++r) { Vst[n][r] = 0.f; Ist[n][r] = 0.f; Zac[n][r] = 0.f; }

  // z buffer 0 must be zero at t=0 (z carry starts at 0)
  for (int i2 = tid; i2 < TB * HID; i2 += 128) sm_z[i2] = (_Float16)0;

  const unsigned xbase = (unsigned)(size_t)(void*)&sm_x[0];  // LDS byte offset

  if (tid < 32) tdm_load_tile(x16 + (size_t)rb * INF, xbase);  // tile t=0
  __syncthreads();

  for (int t = 0; t < TT; ++t) {
    const int pb = t & 1;
    if (tid < 32) __builtin_amdgcn_s_wait_tensorcnt(0);  // tile t landed
    __syncthreads();  // tile t visible; prev-step z writes done & reads drained
    if (tid < 32 && (t + 1) < TT)                        // prefetch tile t+1
      tdm_load_tile(x16 + ((size_t)(t + 1) * BB + rb) * INF,
                    xbase + (unsigned)(((t + 1) & 1) * (TB * INF * 2)));

    // A fragments from LDS
    const _Float16* xr = sm_x + pb * TB * INF + lrow * INF;
    const _Float16* zr = sm_z + pb * TB * HID + lrow * HID;
    v16h ax[2], az[4];
#pragma unroll
    for (int c = 0; c < 2; ++c) ax[c] = load_fragA(xr + 32 * c + 16 * lhi);
#pragma unroll
    for (int c = 0; c < 4; ++c) az[c] = load_fragA(zr + 32 * c + 16 * lhi);

    _Float16* zw = sm_z + (pb ^ 1) * TB * HID;

#pragma unroll
    for (int n = 0; n < 2; ++n) {
      v8f cur = {0.f, 0.f, 0.f, 0.f, 0.f, 0.f, 0.f, 0.f};
#pragma unroll
      for (int c = 0; c < 2; ++c) cur = wmma_f16(ax[c], bcomb[n][c], cur);
      v8f rec = {0.f, 0.f, 0.f, 0.f, 0.f, 0.f, 0.f, 0.f};
#pragma unroll
      for (int c = 0; c < 4; ++c) rec = wmma_f16(az[c], brec[n][c], rec);

      const int col = 32 * wave + 16 * n + lrow;
#pragma unroll
      for (int r = 0; r < 8; ++r) {
        float vv = Vst[n][r], ii = Ist[n][r];
        float vd = vv + dtm[n] * ((vleak - vv) + ii);   // membrane decay
        float id = ii - dts[n] * ii;                    // synapse decay
        bool  sp = (vd - vth[n]) > 0.0f;                // heaviside fwd
        float z  = sp ? 1.0f : 0.0f;
        Vst[n][r] = sp ? vrst[n] : vd;                  // reset
        Ist[n][r] = id + cur[r] + rec[r];               // inject currents
        Zac[n][r] += z;
        zw[(r + 8 * lhi) * HID + col] = (_Float16)z;    // z for next step
      }
    }
  }

  // ---- finalize: z_sum = Zacc @ W_out.T + T*b_out ; spikerate partial ----
  __syncthreads();
#pragma unroll
  for (int n = 0; n < 2; ++n) {
    const int col = 32 * wave + 16 * n + lrow;
#pragma unroll
    for (int r = 0; r < 8; ++r) sm_zs[(r + 8 * lhi) * HID + col] = Zac[n][r];
  }
  __syncthreads();

  if (tid < 32) {
    const int row = tid >> 1, o = tid & 1;
    float acc = 0.0f;
    for (int h = 0; h < HID; ++h) acc += sm_zs[row * HID + h] * W_out[o * HID + h];
    out[(rb + row) * 2 + o] = acc + (float)TT * b_out[o];
  }
  if (tid == 0) {
    float s = 0.0f;
    for (int i2 = 0; i2 < TB * HID; ++i2) s += sm_zs[i2];
    spk_partial[blockIdx.x] = s;   // deterministic per-WG partial (no atomics)
  }
}

// ------------------- kernel 4: deterministic spikerate ---------------------
__global__ void finish_kernel(const float* __restrict__ partials,
                              float* __restrict__ out) {
  if (threadIdx.x == 0 && blockIdx.x == 0) {
    float s = 0.0f;
    for (int i = 0; i < NWG; ++i) s += partials[i];   // fixed order
    out[BB * 2] = s / (float)((size_t)TT * BB * HID);
  }
}

// ---------------------------------------------------------------------------
extern "C" void kernel_launch(void* const* d_in, const int* in_sizes, int n_in,
                              void* d_out, int out_size, void* d_ws,
                              size_t ws_size, hipStream_t stream) {
  (void)in_sizes; (void)n_in; (void)out_size; (void)ws_size;
  const float* x       = (const float*)d_in[0];
  const float* W_in    = (const float*)d_in[1];
  const float* cW_in   = (const float*)d_in[2];
  const float* cW_rec  = (const float*)d_in[3];
  const float* W_out   = (const float*)d_in[4];
  const float* b_out   = (const float*)d_in[5];
  /* d_in[6] = alpha: unused in forward (surrogate grad only) */
  const float* v_th    = (const float*)d_in[7];
  const float* v_leak  = (const float*)d_in[8];
  const float* v_reset = (const float*)d_in[9];
  const float* tau_mem = (const float*)d_in[10];
  const float* tau_syn = (const float*)d_in[11];

  char* ws = (char*)d_ws;
  _Float16* x16   = (_Float16*)ws;                               // 64 MiB
  _Float16* wcomb = (_Float16*)(ws + (size_t)TT * BB * INF * 2); // 16 KiB
  _Float16* wrec  = (_Float16*)((char*)wcomb + HID * INF * 2);   // 32 KiB
  float* partials = (float*)((char*)wrec + HID * HID * 2);       // 128 B

  float* out = (float*)d_out;

  const int NX = TT * BB * INF;  // 2^25 halves, divisible by 1024
  convx_kernel<<<NX / (256 * 4), 256, 0, stream>>>(x, x16);
  prep_kernel<<<1, HID, 0, stream>>>(W_in, cW_in, cW_rec, wcomb, wrec);
  snn_scan_kernel<<<NWG, 128, 0, stream>>>(x16, wcomb, wrec, W_out, b_out, v_th,
                                           v_leak, v_reset, tau_mem, tau_syn,
                                           out, partials);
  finish_kernel<<<1, 32, 0, stream>>>(partials, out);
}
